// SetUpconvModule_14620068676155
// MI455X (gfx1250) — compile-verified, matrix operand order checked by reading, not run on hardware
//
#include <hip/hip_runtime.h>

// ---------------------------------------------------------------------------
// SetUpconv (PointNet++ upconv) for gfx1250 / MI455X.
//   K1: brute-force KNN (K=8) with xyz2 staged in LDS, top-8 in registers.
//   K2: fused gather -> MLP0 -> MLP1 -> maxpool(K) -> concat(feat1) -> MLP2,
//       all GEMMs on v_wmma_f32_16x16x32_bf16 (bf16 in, f32 accumulate),
//       bias folded into accumulator init, ReLU fused on writeback.
// ---------------------------------------------------------------------------

#define B_   8
#define N1_  8192
#define N2_  2048
#define KNN_ 8
#define C1_  64
#define C2_  128
#define M0_  128   // MLP0 out
#define M1_  128   // MLP1 out
#define M2_  128   // MLP2 out
#define K1PAD 160  // 131 padded to 5 chunks of 32
#define K3_  192   // 128 + 64

typedef __attribute__((ext_vector_type(16))) __bf16 v16bf;
typedef __attribute__((ext_vector_type(8)))  __bf16 v8bf;
typedef __attribute__((ext_vector_type(8)))  float  v8f;

__device__ __forceinline__ v8f wmma_bf16(v16bf a, v16bf b, v8f c) {
  // (neg_a, A, neg_b, B, c_mod, C, reuse_a, reuse_b)
  return __builtin_amdgcn_wmma_f32_16x16x32_bf16(false, a, false, b,
                                                 (short)0, c, false, false);
}

// A-fragment (16x32 bf16) from LDS, row-major with stride `stride` (bf16 units).
// Layout per ISA: lane<16 -> row=base_row, Ks {k0..k0+7, k0+16..k0+23};
//                 lane>=16 -> Ks shifted by +8.  koff already encodes the +8.
__device__ __forceinline__ v16bf load_a_lds(const __bf16* base, int stride,
                                            int row, int kbase) {
  const __bf16* p = base + row * stride + kbase;
  v8bf lo = *reinterpret_cast<const v8bf*>(p);
  v8bf hi = *reinterpret_cast<const v8bf*>(p + 16);
  v16bf a;
#pragma unroll
  for (int i = 0; i < 8; ++i) { a[i] = lo[i]; a[8 + i] = hi[i]; }
  return a;
}

// B-fragment (32x16 bf16) from a row-major float weight W[K][N]; column n,
// Ks {kbase..kbase+7, kbase+16..kbase+23}; zero-pad beyond klim.
__device__ __forceinline__ v16bf load_b_global(const float* __restrict__ W,
                                               int N, int klim, int kbase,
                                               int n) {
  v16bf b;
#pragma unroll
  for (int i = 0; i < 8; ++i) {
    int k0 = kbase + i;
    int k1 = kbase + 16 + i;
    b[i]     = (k0 < klim) ? (__bf16)W[k0 * N + n] : (__bf16)0.0f;
    b[8 + i] = (k1 < klim) ? (__bf16)W[k1 * N + n] : (__bf16)0.0f;
  }
  return b;
}

// ---------------------------------------------------------------------------
// Kernel 1: KNN.  grid = (N1/256, B), block = 256.
// ---------------------------------------------------------------------------
__global__ __launch_bounds__(256) void knn_kernel(
    const float* __restrict__ xyz1, const float* __restrict__ xyz2,
    int* __restrict__ idxout) {
  __shared__ __align__(16) float sx[N2_ * 3];  // 24 KB

  const int tid = threadIdx.x;
  const int b   = blockIdx.y;

  const float4* src4 = reinterpret_cast<const float4*>(xyz2 + (size_t)b * N2_ * 3);
  float4* dst4 = reinterpret_cast<float4*>(sx);
#pragma unroll
  for (int t = tid; t < (N2_ * 3) / 4; t += 256) dst4[t] = src4[t];
  __syncthreads();

  const int i = blockIdx.x * 256 + tid;
  const float* q = xyz1 + ((size_t)b * N1_ + i) * 3;
  const float qx = q[0], qy = q[1], qz = q[2];

  float dk[KNN_];
  int   ik[KNN_];
#pragma unroll
  for (int t = 0; t < KNN_; ++t) { dk[t] = 3.0e38f; ik[t] = 0; }

  for (int j = 0; j < N2_; ++j) {
    float dx = sx[j * 3 + 0] - qx;
    float dy = sx[j * 3 + 1] - qy;
    float dz = sx[j * 3 + 2] - qz;
    float d2 = dx * dx + dy * dy + dz * dz;
    if (d2 < dk[KNN_ - 1]) {
      dk[KNN_ - 1] = d2; ik[KNN_ - 1] = j;
#pragma unroll
      for (int t = KNN_ - 1; t > 0; --t) {
        if (dk[t] < dk[t - 1]) {
          float td = dk[t]; dk[t] = dk[t - 1]; dk[t - 1] = td;
          int   ti = ik[t]; ik[t] = ik[t - 1]; ik[t - 1] = ti;
        }
      }
    }
  }

  int* o = idxout + ((size_t)b * N1_ + i) * KNN_;
#pragma unroll
  for (int t = 0; t < KNN_; ++t) o[t] = ik[t];
}

// ---------------------------------------------------------------------------
// Kernel 2: fused gather + MLP0 + MLP1 + maxpool + MLP2.
// grid = (N1/16, B), block = 256 (8 wave32s; wave w owns output N-tile w).
// ---------------------------------------------------------------------------
__global__ __launch_bounds__(256) void fused_kernel(
    const float* __restrict__ xyz1, const float* __restrict__ xyz2,
    const float* __restrict__ feat1, const float* __restrict__ feat2,
    const float* __restrict__ W0, const float* __restrict__ b0,
    const float* __restrict__ W1, const float* __restrict__ b1,
    const float* __restrict__ W2, const float* __restrict__ b2,
    const int* __restrict__ idxbuf, float* __restrict__ out) {
  // Abuf holds, in sequence: A (128x160 bf16), H1 (128x128), H2 (128x128).
  __shared__ __align__(16) __bf16 Abuf[128 * K1PAD];  // 40 KB
  __shared__ __align__(16) __bf16 A3[16 * K3_];       // 6 KB
  __shared__ int   sidx[128];
  __shared__ float sq[16 * 3];

  const int tid  = threadIdx.x;
  const int lane = tid & 31;
  const int wv   = tid >> 5;            // 0..7 = output N-tile
  const int b    = blockIdx.y;
  const int i0   = blockIdx.x * 16;

  // ---- stage 0: neighbor indices + query coords
  if (tid < 128)
    sidx[tid] = idxbuf[((size_t)b * N1_ + i0 + (tid >> 3)) * KNN_ + (tid & 7)];
  if (tid < 48)
    sq[tid] = xyz1[((size_t)b * N1_ + i0 + tid / 3) * 3 + (tid % 3)];
  __syncthreads();

  // ---- stage 1: gather A = [feat2_g | xyz_diff | 0-pad] as bf16 rows of 160
  {
    const int r = tid >> 1, h = tid & 1;  // 2 threads per row
    const int p = r >> 3;
    const int src = sidx[r];
    const float* f2 = feat2 + ((size_t)b * N2_ + src) * C2_;
    __bf16* arow = Abuf + r * K1PAD;
    const int c0 = h * 64;
#pragma unroll
    for (int c = 0; c < 64; c += 4) {
      float4 v = *reinterpret_cast<const float4*>(f2 + c0 + c);
      arow[c0 + c + 0] = (__bf16)v.x;
      arow[c0 + c + 1] = (__bf16)v.y;
      arow[c0 + c + 2] = (__bf16)v.z;
      arow[c0 + c + 3] = (__bf16)v.w;
    }
    if (h) {
      const float* x2 = xyz2 + ((size_t)b * N2_ + src) * 3;
#pragma unroll
      for (int d = 0; d < 3; ++d)
        arow[C2_ + d] = (__bf16)(x2[d] - sq[p * 3 + d]);
#pragma unroll
      for (int c = C2_ + 3; c < K1PAD; ++c) arow[c] = (__bf16)0.0f;
    }
  }
  __syncthreads();

  const int n     = wv * 16 + (lane & 15);      // this lane's output column
  const int koff  = (lane & 16) ? 8 : 0;        // K shift for upper half-wave
  const int arow0 = lane & 15;                  // A row within M-slab
  const int moff  = (lane & 16) ? 8 : 0;        // D-row offset for this lane

  // ---- stage 2: GEMM1  A(128x131) @ W0(131x128) + b0, ReLU -> H1
  v16bf Bf[6];
#pragma unroll
  for (int kc = 0; kc < 5; ++kc)
    Bf[kc] = load_b_global(W0, M0_, C2_ + 3, kc * 32 + koff, n);

  v8f acc[8];
  {
    const float bias = b0[n];
#pragma unroll
    for (int s = 0; s < 8; ++s)
#pragma unroll
      for (int v = 0; v < 8; ++v) acc[s][v] = bias;
  }
#pragma unroll
  for (int s = 0; s < 8; ++s)
#pragma unroll
    for (int kc = 0; kc < 5; ++kc)
      acc[s] = wmma_bf16(load_a_lds(Abuf, K1PAD, s * 16 + arow0, kc * 32 + koff),
                         Bf[kc], acc[s]);
  __syncthreads();  // all A reads done before overwrite
#pragma unroll
  for (int s = 0; s < 8; ++s)
#pragma unroll
    for (int v = 0; v < 8; ++v) {
      float x = acc[s][v];
      Abuf[(s * 16 + moff + v) * M0_ + n] = (__bf16)(x > 0.0f ? x : 0.0f);
    }
  __syncthreads();

  // ---- stage 3: GEMM2  H1(128x128) @ W1(128x128) + b1, ReLU -> H2
#pragma unroll
  for (int kc = 0; kc < 4; ++kc)
    Bf[kc] = load_b_global(W1, M1_, M0_, kc * 32 + koff, n);
  {
    const float bias = b1[n];
#pragma unroll
    for (int s = 0; s < 8; ++s)
#pragma unroll
      for (int v = 0; v < 8; ++v) acc[s][v] = bias;
  }
#pragma unroll
  for (int s = 0; s < 8; ++s)
#pragma unroll
    for (int kc = 0; kc < 4; ++kc)
      acc[s] = wmma_bf16(load_a_lds(Abuf, M0_, s * 16 + arow0, kc * 32 + koff),
                         Bf[kc], acc[s]);
  __syncthreads();
#pragma unroll
  for (int s = 0; s < 8; ++s)
#pragma unroll
    for (int v = 0; v < 8; ++v) {
      float x = acc[s][v];
      Abuf[(s * 16 + moff + v) * M1_ + n] = (__bf16)(x > 0.0f ? x : 0.0f);
    }
  __syncthreads();

  // ---- stage 4: maxpool over KNN_ neighbors + concat feat1 -> A3 (16x192)
  {
    const int p  = tid >> 4;         // point 0..15
    const int nb = (tid & 15) * 8;   // 8-col block
#pragma unroll
    for (int c = 0; c < 8; ++c) {
      float m = (float)Abuf[(p * 8) * M1_ + nb + c];
#pragma unroll
      for (int j = 1; j < KNN_; ++j) {
        float x = (float)Abuf[(p * 8 + j) * M1_ + nb + c];
        m = x > m ? x : m;
      }
      A3[p * K3_ + nb + c] = (__bf16)m;
    }
    if (tid < 128) {
      const int pp = tid >> 3, cb = (tid & 7) * 8;
      const float* f1 = feat1 + ((size_t)b * N1_ + i0 + pp) * C1_ + cb;
#pragma unroll
      for (int c = 0; c < 8; ++c)
        A3[pp * K3_ + M1_ + cb + c] = (__bf16)f1[c];
    }
  }
  __syncthreads();

  // ---- stage 5: GEMM3  A3(16x192) @ W2(192x128) + b2, ReLU -> out
  v8f acc3;
  {
    const float bias = b2[n];
#pragma unroll
    for (int v = 0; v < 8; ++v) acc3[v] = bias;
  }
#pragma unroll
  for (int kc = 0; kc < 6; ++kc)
    acc3 = wmma_bf16(load_a_lds(A3, K3_, arow0, kc * 32 + koff),
                     load_b_global(W2, M2_, K3_, kc * 32 + koff, n), acc3);
#pragma unroll
  for (int v = 0; v < 8; ++v) {
    float x = acc3[v];
    out[((size_t)b * N1_ + i0 + moff + v) * M2_ + n] = x > 0.0f ? x : 0.0f;
  }
}

// ---------------------------------------------------------------------------
extern "C" void kernel_launch(void* const* d_in, const int* in_sizes, int n_in,
                              void* d_out, int out_size, void* d_ws,
                              size_t ws_size, hipStream_t stream) {
  const float* xyz1  = (const float*)d_in[0];
  const float* xyz2  = (const float*)d_in[1];
  const float* feat1 = (const float*)d_in[2];
  const float* feat2 = (const float*)d_in[3];
  const float* W0 = (const float*)d_in[4];
  const float* b0 = (const float*)d_in[5];
  const float* W1 = (const float*)d_in[6];
  const float* b1 = (const float*)d_in[7];
  const float* W2 = (const float*)d_in[8];
  const float* b2 = (const float*)d_in[9];
  float* out  = (float*)d_out;
  int* idxbuf = (int*)d_ws;  // B*N1*K ints = 2 MB

  knn_kernel<<<dim3(N1_ / 256, B_), 256, 0, stream>>>(xyz1, xyz2, idxbuf);
  fused_kernel<<<dim3(N1_ / 16, B_), 256, 0, stream>>>(
      xyz1, xyz2, feat1, feat2, W0, b0, W1, b1, W2, b2, idxbuf, out);
}
